// VQ_58342835748939
// MI455X (gfx1250) — matrix operand order repaired and checked by reading, not compiled, not run on hardware
//
#include <hip/hip_runtime.h>

// ---------------------------------------------------------------------------
// VQ-VAE pairwise L2 distance:  out[n][k] = sqrt(max(||x_n||^2 + ||e_k||^2
//                                                    - 2 x_n . e_k, 0))
// N=65536 rows, K=1024 codebook entries, D=64.
// Strategy: bf16 split-compensated GEMM on v_wmma_f32_16x16x32_bf16,
// codebook staged in LDS as (hi, lo) bf16 pairs, epilogue fused.
// Output (268 MB f32) is the bandwidth bottleneck (~11.5 us at 23.3 TB/s);
// epilogue uses raw v_sqrt_f32 to keep VALU from contending with WMMA issue.
// ---------------------------------------------------------------------------

typedef __attribute__((ext_vector_type(16))) __bf16         v16bf;
typedef __attribute__((ext_vector_type(16))) unsigned short v16u;
typedef __attribute__((ext_vector_type(8)))  float          v8f;

static constexpr int N_ROWS   = 65536;
static constexpr int K_COLS   = 1024;
static constexpr int DIM      = 64;

static constexpr int BLK_ROWS = 128;   // 8 waves x 16 rows
static constexpr int BLK_COLS = 512;   // codebook slice staged in LDS
static constexpr int THREADS  = 256;   // 8 wave32

// Dynamic LDS layout (bytes):
static constexpr int LDS_EHI  = 0;                                   // ushort[BLK_COLS*DIM]
static constexpr int LDS_ELO  = LDS_EHI + BLK_COLS * DIM * 2;        // ushort[BLK_COLS*DIM]
static constexpr int LDS_ESQ  = LDS_ELO + BLK_COLS * DIM * 2;        // float[BLK_COLS]
static constexpr int LDS_XSQ  = LDS_ESQ + BLK_COLS * 4;              // float[BLK_ROWS]
static constexpr int LDS_SIZE = LDS_XSQ + BLK_ROWS * 4;              // ~130.5 KB -> 2 blocks/WGP

// round-to-nearest-even f32 -> bf16 bits
__device__ __forceinline__ unsigned short bf16_rne(float f) {
    unsigned u = __builtin_bit_cast(unsigned, f);
    unsigned r = 0x7FFFu + ((u >> 16) & 1u);
    return (unsigned short)((u + r) >> 16);
}

// split f32 into hi (truncated bf16, exactly representable) + lo (residual)
__device__ __forceinline__ void split2(float f, unsigned short& h, unsigned short& l) {
    unsigned u  = __builtin_bit_cast(unsigned, f);
    float    hf = __builtin_bit_cast(float, u & 0xFFFF0000u);
    h = (unsigned short)(u >> 16);
    l = bf16_rne(f - hf);
}

__global__ __launch_bounds__(THREADS) void vq_dist_kernel(
    const float* __restrict__ x,     // [N_ROWS, DIM]
    const float* __restrict__ e,     // [K_COLS, DIM]
    float* __restrict__ out)         // [N_ROWS, K_COLS]
{
    extern __shared__ char smem[];
    unsigned short* ehi = (unsigned short*)(smem + LDS_EHI);
    unsigned short* elo = (unsigned short*)(smem + LDS_ELO);
    float*          esq = (float*)(smem + LDS_ESQ);
    float*          xsq = (float*)(smem + LDS_XSQ);

    const int tid      = threadIdx.x;
    const int row_base = blockIdx.x * BLK_ROWS;
    const int col_base = blockIdx.y * BLK_COLS;

    // ---- Phase 1: stage codebook slice into LDS as bf16 hi/lo, plus ||e||^2
    for (int r = tid; r < BLK_COLS; r += THREADS) {
        const float* src = e + (size_t)(col_base + r) * DIM;
        unsigned short* dh = ehi + r * DIM;
        unsigned short* dl = elo + r * DIM;
        float acc = 0.0f;
        #pragma unroll
        for (int k = 0; k < DIM; k += 4) {
            float4 v = *(const float4*)(src + k);
            unsigned short h, l;
            split2(v.x, h, l); dh[k + 0] = h; dl[k + 0] = l; acc = fmaf(v.x, v.x, acc);
            split2(v.y, h, l); dh[k + 1] = h; dl[k + 1] = l; acc = fmaf(v.y, v.y, acc);
            split2(v.z, h, l); dh[k + 2] = h; dl[k + 2] = l; acc = fmaf(v.z, v.z, acc);
            split2(v.w, h, l); dh[k + 3] = h; dl[k + 3] = l; acc = fmaf(v.w, v.w, acc);
        }
        esq[r] = acc;
    }
    // ---- ||x||^2 for this block's rows
    for (int r = tid; r < BLK_ROWS; r += THREADS) {
        const float* src = x + (size_t)(row_base + r) * DIM;
        float acc = 0.0f;
        #pragma unroll
        for (int k = 0; k < DIM; k += 4) {
            float4 v = *(const float4*)(src + k);
            acc = fmaf(v.x, v.x, acc);
            acc = fmaf(v.y, v.y, acc);
            acc = fmaf(v.z, v.z, acc);
            acc = fmaf(v.w, v.w, acc);
        }
        xsq[r] = acc;
    }
    __syncthreads();

    // ---- Phase 2: per-wave A fragments (16 rows x 64 K), hi + lo
    const int wave = tid >> 5;          // 0..7
    const int lane = tid & 31;
    const int half = lane >> 4;         // 0: lanes 0-15, 1: lanes 16-31
    const int m    = lane & 15;

    // A fragment, contiguous K order (same as B): lane holds row m; elements 0..7 = K
    // kA..kA+7, elements 8..15 = K kB..kB+7 (k = 16*half + s per 32-K chunk).
    const float* xrow = x + (size_t)(row_base + wave * 16 + m) * DIM;
    v16u ahi_u[2], alo_u[2];
    #pragma unroll
    for (int c = 0; c < 2; ++c) {
        const int kA = c * 32 + half * 16;
        const int kB = c * 32 + half * 16 + 8;
        #pragma unroll
        for (int j = 0; j < 8; ++j) {
            unsigned short h, l;
            split2(xrow[kA + j], h, l);
            ahi_u[c][j] = h;     alo_u[c][j] = l;
            split2(xrow[kB + j], h, l);
            ahi_u[c][8 + j] = h; alo_u[c][8 + j] = l;
        }
    }
    const v16bf a_hi0 = __builtin_bit_cast(v16bf, ahi_u[0]);
    const v16bf a_hi1 = __builtin_bit_cast(v16bf, ahi_u[1]);
    const v16bf a_lo0 = __builtin_bit_cast(v16bf, alo_u[0]);
    const v16bf a_lo1 = __builtin_bit_cast(v16bf, alo_u[1]);

    // Hoist ||x||^2 for the 8 D-matrix rows this lane touches:
    // D layout: VGPR r <-> M = r + 8*half, N = m.
    float xs[8];
    #pragma unroll
    for (int r = 0; r < 8; ++r)
        xs[r] = xsq[wave * 16 + 8 * half + r];

    // ---- Phase 3: sweep 16-column tiles of the staged codebook
    for (int t = 0; t < BLK_COLS / 16; ++t) {
        const int colL = t * 16 + m;   // block-local column (= codebook row)
        // B fragment, contiguous K order (same as A): lanes 0-15 hold K 0..15 of column m,
        // lanes 16-31 hold K 16..31; codebook is row-major so this is a
        // contiguous 32B LDS read per lane, per 32-K chunk.
        const unsigned short* bh = ehi + colL * DIM + half * 16;
        const unsigned short* bl = elo + colL * DIM + half * 16;
        const v16bf b_hi0 = *(const v16bf*)(bh);
        const v16bf b_hi1 = *(const v16bf*)(bh + 32);
        const v16bf b_lo0 = *(const v16bf*)(bl);
        const v16bf b_lo1 = *(const v16bf*)(bl + 32);

        v8f c = {};
        // hi*hi (K=0..63)
        c = __builtin_amdgcn_wmma_f32_16x16x32_bf16(false, a_hi0, false, b_hi0, (short)0, c, false, false);
        c = __builtin_amdgcn_wmma_f32_16x16x32_bf16(false, a_hi1, false, b_hi1, (short)0, c, false, false);
        // lo*hi
        c = __builtin_amdgcn_wmma_f32_16x16x32_bf16(false, a_lo0, false, b_hi0, (short)0, c, false, false);
        c = __builtin_amdgcn_wmma_f32_16x16x32_bf16(false, a_lo1, false, b_hi1, (short)0, c, false, false);
        // hi*lo
        c = __builtin_amdgcn_wmma_f32_16x16x32_bf16(false, a_hi0, false, b_lo0, (short)0, c, false, false);
        c = __builtin_amdgcn_wmma_f32_16x16x32_bf16(false, a_hi1, false, b_lo1, (short)0, c, false, false);

        // ---- fused epilogue: sqrt(max(xsq + esq - 2*cross, 0))
        // Raw v_sqrt_f32 (~1 ulp) instead of the IEEE-precise expansion:
        // keeps the loop WMMA/store bound instead of VALU bound.
        const float es = esq[colL];
        const size_t col = (size_t)(col_base + colL);
        #pragma unroll
        for (int r = 0; r < 8; ++r) {
            const int rowL = wave * 16 + 8 * half + r;
            float d = fmaf(-2.0f, c[r], xs[r] + es);
            d = __builtin_amdgcn_sqrtf(fmaxf(d, 0.0f));
            out[(size_t)(row_base + rowL) * K_COLS + col] = d;
        }
    }
}

extern "C" void kernel_launch(void* const* d_in, const int* in_sizes, int n_in,
                              void* d_out, int out_size, void* d_ws, size_t ws_size,
                              hipStream_t stream) {
    const float* x = (const float*)d_in[0];   // [65536, 64]
    const float* e = (const float*)d_in[1];   // [1024, 64]
    float* out     = (float*)d_out;           // [65536, 1024]
    (void)in_sizes; (void)n_in; (void)out_size; (void)d_ws; (void)ws_size;

    hipFuncSetAttribute((const void*)vq_dist_kernel,
                        hipFuncAttributeMaxDynamicSharedMemorySize, LDS_SIZE);

    dim3 grid(N_ROWS / BLK_ROWS, K_COLS / BLK_COLS);   // 512 x 2
    vq_dist_kernel<<<grid, THREADS, LDS_SIZE, stream>>>(x, e, out);
}
